// GATLayer_2654289789412
// MI455X (gfx1250) — matrix operand order repaired and checked by reading, not compiled, not run on hardware
//
#include <hip/hip_runtime.h>

typedef __attribute__((ext_vector_type(16))) __bf16 v16bf;
typedef __attribute__((ext_vector_type(8)))  float  v8f;

#define IN_DIM  128
#define OUT_DIM 64
#define NEG_SLOPE 0.01f

// ---------- monotonic float<->uint encoding for atomic max over floats ----------
__device__ __forceinline__ unsigned fenc(float f) {
  unsigned u = __float_as_uint(f);
  return (u & 0x80000000u) ? ~u : (u | 0x80000000u);
}
__device__ __forceinline__ float fdec(unsigned u) {
  u = (u & 0x80000000u) ? (u & 0x7FFFFFFFu) : ~u;
  return __uint_as_float(u);
}

// ---------- kernel 1: z = h @ W via bf16 WMMA ----------
// block = 256 threads = 8 waves; each wave computes a 16-row x 64-col strip.
// W is pre-packed in LDS in the exact per-lane WMMA B-operand layout:
//   Bl[kt][ct][lane][i] = W[kt*32 + (lane>>4)*8 + (i&7) + (i>>3)*16][ct*16 + (lane&15)]
// so each B operand is one 32B contiguous, 32B-aligned LDS read per lane.
__global__ __launch_bounds__(256) void gat_gemm_wmma(
    const float* __restrict__ h, const float* __restrict__ W,
    float* __restrict__ z, int N) {
  __shared__ __align__(32) __bf16 Bl[4][4][32][16]; // 16 KB

  // pack: 8192 elements, 32 per thread (gather from L2-resident 32KB W)
#pragma unroll
  for (int k = 0; k < 32; ++k) {
    int f    = threadIdx.x + k * 256;
    int i    = f & 15;
    int ln   = (f >> 4) & 31;
    int ct   = (f >> 9) & 3;
    int kt   = f >> 11;
    int row  = kt * 32 + (ln >> 4) * 8 + (i & 7) + (i >> 3) * 16;
    int col  = ct * 16 + (ln & 15);
    Bl[kt][ct][ln][i] = (__bf16)W[row * OUT_DIM + col];
  }
  __syncthreads();

  const int wave     = threadIdx.x >> 5;
  const int lane     = threadIdx.x & 31;
  const int rowLocal = lane & 15;
  const int half     = lane >> 4;          // 0: lanes 0-15, 1: lanes 16-31
  const int tileRow  = blockIdx.x * 128 + wave * 16;

  int row  = tileRow + rowLocal;
  int rowC = row < N ? row : (N - 1);      // clamp loads; keep EXEC all-ones
  const float* hrow = h + (size_t)rowC * IN_DIM;

  v8f acc[4];
#pragma unroll
  for (int ct = 0; ct < 4; ++ct) acc[ct] = (v8f){};

#pragma unroll
  for (int kt = 0; kt < 4; ++kt) {         // K-steps of 32
    // A tile 16x32, ISA layout: lane half selects K 0-7/8-15 and 16-23/24-31
    const int kbase = kt * 32 + half * 8;
    float4 f0 = *(const float4*)(hrow + kbase);
    float4 f1 = *(const float4*)(hrow + kbase + 4);
    float4 f2 = *(const float4*)(hrow + kbase + 16);
    float4 f3 = *(const float4*)(hrow + kbase + 20);
    v16bf a;
    a[0]  = (__bf16)f0.x; a[1]  = (__bf16)f0.y; a[2]  = (__bf16)f0.z; a[3]  = (__bf16)f0.w;
    a[4]  = (__bf16)f1.x; a[5]  = (__bf16)f1.y; a[6]  = (__bf16)f1.z; a[7]  = (__bf16)f1.w;
    a[8]  = (__bf16)f2.x; a[9]  = (__bf16)f2.y; a[10] = (__bf16)f2.z; a[11] = (__bf16)f2.w;
    a[12] = (__bf16)f3.x; a[13] = (__bf16)f3.y; a[14] = (__bf16)f3.z; a[15] = (__bf16)f3.w;

    // batch all 4 B tiles first so the WMMAs issue back-to-back (no per-op dscnt wait)
    v16bf b[4];
#pragma unroll
    for (int ct = 0; ct < 4; ++ct)
      b[ct] = *(const v16bf*)&Bl[kt][ct][lane][0];  // 2x ds_load_b128 each

#pragma unroll
    for (int ct = 0; ct < 4; ++ct)
      acc[ct] = __builtin_amdgcn_wmma_f32_16x16x32_bf16(
          false, a, false, b[ct], (short)0, acc[ct], false, false);
  }

  // D layout: VGPR r -> row (r + 8*half), col = lane&15
#pragma unroll
  for (int ct = 0; ct < 4; ++ct) {
#pragma unroll
    for (int r = 0; r < 8; ++r) {
      int m = tileRow + half * 8 + r;
      if (m < N) z[(size_t)m * OUT_DIM + ct * 16 + rowLocal] = acc[ct][r];
    }
  }
}

// ---------- kernel 2: s = z@a_l, t = z@a_r (one wave per node) ----------
__global__ __launch_bounds__(256) void gat_st(
    const float* __restrict__ z, const float* __restrict__ al,
    const float* __restrict__ ar, float* __restrict__ s,
    float* __restrict__ t, int N) {
  int tid  = blockIdx.x * 256 + threadIdx.x;
  int n    = tid >> 5;
  int lane = tid & 31;
  if (n >= N) return;
  float2 zz = *(const float2*)(z + (size_t)n * OUT_DIM + lane * 2);
  float2 l  = *(const float2*)(al + lane * 2);
  float2 r  = *(const float2*)(ar + lane * 2);
  float ps = zz.x * l.x + zz.y * l.y;
  float pt = zz.x * r.x + zz.y * r.y;
#pragma unroll
  for (int off = 16; off > 0; off >>= 1) {
    ps += __shfl_down(ps, off, 32);
    pt += __shfl_down(pt, off, 32);
  }
  if (lane == 0) { s[n] = ps; t[n] = pt; }
}

// ---------- kernel 3: init out / m / denom ----------
__global__ void gat_init(float* __restrict__ out, unsigned* __restrict__ menc,
                         float* __restrict__ denom, int N) {
  int i = blockIdx.x * blockDim.x + threadIdx.x;
  if (i < N * OUT_DIM) out[i] = 0.0f;
  if (i < N) { menc[i] = 0u; denom[i] = 0.0f; }
}

// ---------- kernel 4: e = lrelu(s[src]+t[dst]); segment max via atomicMax ----------
__global__ void gat_edge_max(const int* __restrict__ src, const int* __restrict__ dst,
                             const float* __restrict__ s, const float* __restrict__ t,
                             float* __restrict__ ebuf, unsigned* __restrict__ menc,
                             int E) {
  int i = blockIdx.x * blockDim.x + threadIdx.x;
  if (i >= E) return;
  float e = s[src[i]] + t[dst[i]];
  e = e > 0.0f ? e : NEG_SLOPE * e;
  ebuf[i] = e;
  atomicMax(&menc[dst[i]], fenc(e));
}

// ---------- kernel 5: ex = exp(e - m[dst]); segment sum ----------
__global__ void gat_edge_exp(const int* __restrict__ dst,
                             const float* __restrict__ ebuf,
                             const unsigned* __restrict__ menc,
                             float* __restrict__ exbuf,
                             float* __restrict__ denom, int E) {
  int i = blockIdx.x * blockDim.x + threadIdx.x;
  if (i >= E) return;
  int d = dst[i];
  float ex = __expf(ebuf[i] - fdec(menc[d]));
  exbuf[i] = ex;
  atomicAdd(&denom[d], ex);
}

// ---------- kernel 6: out[dst] += (ex/denom[dst]) * z[src]  (one wave/edge) ----------
__global__ __launch_bounds__(256) void gat_edge_scatter(
    const int* __restrict__ src, const int* __restrict__ dst,
    const float* __restrict__ exbuf, const float* __restrict__ denom,
    const float* __restrict__ z, float* __restrict__ out, int E) {
  int tid  = blockIdx.x * 256 + threadIdx.x;
  int e    = tid >> 5;
  int lane = tid & 31;
  if (e >= E) return;
  int d = dst[e], sN = src[e];
  float w = exbuf[e] / denom[d];
  float2 zz = *(const float2*)(z + (size_t)sN * OUT_DIM + lane * 2);
  float* o = out + (size_t)d * OUT_DIM + lane * 2;
  atomicAdd(o,     w * zz.x);
  atomicAdd(o + 1, w * zz.y);
}

extern "C" void kernel_launch(void* const* d_in, const int* in_sizes, int n_in,
                              void* d_out, int out_size, void* d_ws, size_t ws_size,
                              hipStream_t stream) {
  const float* h  = (const float*)d_in[0];
  const int* src  = (const int*)d_in[1];
  const int* dst  = (const int*)d_in[2];
  const float* W  = (const float*)d_in[3];
  const float* al = (const float*)d_in[4];
  const float* ar = (const float*)d_in[5];
  const int N = in_sizes[0] / IN_DIM;   // 100000
  const int E = in_sizes[1];            // 1600000
  float* out = (float*)d_out;

  // workspace layout
  float*    z     = (float*)d_ws;              // N*64
  float*    s     = z + (size_t)N * OUT_DIM;   // N
  float*    t     = s + N;                     // N
  unsigned* menc  = (unsigned*)(t + N);        // N
  float*    denom = (float*)(menc + N);        // N
  float*    ebuf  = denom + N;                 // E
  float*    exbuf = ebuf + E;                  // E

  // 1) z = h @ W  (bf16 WMMA)
  gat_gemm_wmma<<<(N + 127) / 128, 256, 0, stream>>>(h, W, z, N);
  // 2) s, t
  gat_st<<<(N * 32 + 255) / 256, 256, 0, stream>>>(z, al, ar, s, t, N);
  // 3) init out / m / denom
  gat_init<<<(N * OUT_DIM + 255) / 256, 256, 0, stream>>>(out, menc, denom, N);
  // 4) edge max
  gat_edge_max<<<(E + 255) / 256, 256, 0, stream>>>(src, dst, s, t, ebuf, menc, E);
  // 5) edge exp + denom
  gat_edge_exp<<<(E + 255) / 256, 256, 0, stream>>>(dst, ebuf, menc, exbuf, denom, E);
  // 6) weighted scatter
  gat_edge_scatter<<<(E * 32 + 255) / 256, 256, 0, stream>>>(src, dst, exbuf, denom,
                                                             z, out, E);
}
